// LinearTransformerAnticausalEncoder_42691974922502
// MI455X (gfx1250) — matrix operand order repaired and checked by reading, not compiled, not run on hardware
//
#include <hip/hip_runtime.h>
#include <hip/hip_bf16.h>

typedef _Float16 h16;
typedef __attribute__((ext_vector_type(16))) _Float16 v16h;
typedef __attribute__((ext_vector_type(8)))  float    v8f;

// ---- model constants (match reference) ----
constexpr int BB = 4, LL = 2048, DM = 1024, NH = 16, DH = 64, DFF = 4096, NL = 6;
constexpr int MROWS = BB * LL;            // 8192 token rows
constexpr int CH = 128, NCH = LL / CH;    // attention chunking
constexpr float EPS_ATT = 1e-6f;
constexpr float LN_EPS  = 1e-5f;

// ---------------------------------------------------------------------------
// CDNA5 async global->LDS (ASYNCcnt-tracked, bypasses VGPRs)
// ---------------------------------------------------------------------------
__device__ __forceinline__ void async_b128(unsigned lds_off, const void* gptr) {
    asm volatile("global_load_async_to_lds_b128 %0, %1, off"
                 :: "v"(lds_off), "v"((unsigned long long)(size_t)gptr)
                 : "memory");
}
__device__ __forceinline__ void wait_async0() {
    asm volatile("s_wait_asynccnt 0" ::: "memory");
}
__device__ __forceinline__ unsigned lds_off_of(const void* p) {
    return (unsigned)(size_t)p;   // LDS aperture low 32 bits are the DS offset
}

// ---------------------------------------------------------------------------
// flip along L (anticausal trick): out[b, L-1-l, :] = in[b, l, :]
// ---------------------------------------------------------------------------
__global__ void k_flip_copy(const float* __restrict__ in, float* __restrict__ out) {
    int row = blockIdx.x;
    int b = row / LL, l = row % LL;
    int orow = b * LL + (LL - 1 - l);
    const float* src = in + (size_t)row * DM;
    float* dst = out + (size_t)orow * DM;
    for (int c = threadIdx.x; c < DM; c += blockDim.x) dst[c] = src[c];
}

// ---------------------------------------------------------------------------
// fp32 -> f16 elementwise (8 per thread)
// ---------------------------------------------------------------------------
__global__ void k_f2h(const float* __restrict__ in, h16* __restrict__ out, int n) {
    int i = (blockIdx.x * blockDim.x + threadIdx.x) * 8;
    if (i + 7 < n) {
#pragma unroll
        for (int j = 0; j < 8; ++j) out[i + j] = (h16)in[i + j];
    }
}

// ---------------------------------------------------------------------------
// weight transpose-convert: W fp32 [K][N]  ->  Wt f16 [N][K]
// ---------------------------------------------------------------------------
__global__ void k_wt(const float* __restrict__ W, h16* __restrict__ Wt, int K, int N) {
    __shared__ float tile[32][33];
    int n0 = blockIdx.x * 32, k0 = blockIdx.y * 32;
    int tx = threadIdx.x, ty = threadIdx.y;
#pragma unroll
    for (int j = 0; j < 32; j += 8)
        tile[ty + j][tx] = W[(size_t)(k0 + ty + j) * N + (n0 + tx)];
    __syncthreads();
#pragma unroll
    for (int j = 0; j < 32; j += 8)
        Wt[(size_t)(n0 + ty + j) * K + (k0 + tx)] = (h16)tile[tx][ty + j];
}

// ---------------------------------------------------------------------------
// WMMA GEMM: C[M,N] = epi(A[M,K](f16) @ Bt[N,K](f16)^T + bias[N] (+ resid))
// Block tile 128x256, 8 waves (wave32), wave tile 64x64 = 4x4 wmma accums.
// LDS double-buffered; tiles staged with global_load_async_to_lds_b128 so
// HBM/L2 latency hides behind the WMMA pipe; one barrier per K-step.
// EPI: 0 = none, 1 = elu(x)+1 feature map, 2 = exact gelu
// ---------------------------------------------------------------------------
template <int EPI, typename OutT>
__global__ __launch_bounds__(256) void k_gemm(const h16* __restrict__ A,
                                              const h16* __restrict__ Bt,
                                              const float* __restrict__ bias,
                                              const float* __restrict__ resid,
                                              OutT* __restrict__ C,
                                              int M, int N, int K) {
    constexpr int BM = 128, BN = 256, BK = 32, PAD = 40; // 40 halves = 80B pitch
    constexpr int ASZ = BM * PAD, BSZ = BN * PAD;
    __shared__ __align__(16) h16 As[2 * ASZ];
    __shared__ __align__(16) h16 Bs[2 * BSZ];

    const int tid  = threadIdx.x;
    const int lane = tid & 31;
    const int wave = tid >> 5;
    const int wm = (wave >> 2) * 64;   // wave M offset: 0 / 64
    const int wn = (wave & 3) * 64;    // wave N offset: 0 / 64 / 128 / 192
    const int bm = blockIdx.y * BM, bn = blockIdx.x * BN;
    const int lh   = lane & 15;
    const int ka8  = (lane >> 4) * 8;  // A-frag K base (interleaved layout)
    const int kb16 = (lane >> 4) * 8;  // B-frag K base (interleaved layout)

    const unsigned asBase = lds_off_of(&As[0]);
    const unsigned bsBase = lds_off_of(&Bs[0]);

    v8f acc[4][4];
#pragma unroll
    for (int mi = 0; mi < 4; ++mi)
#pragma unroll
        for (int ni = 0; ni < 4; ++ni)
            acc[mi][ni] = (v8f){0.f, 0.f, 0.f, 0.f, 0.f, 0.f, 0.f, 0.f};

    union FU { uint4 u[2]; v16h v; };

    // stage one 128x32 A tile + 256x32 B tile into LDS buffer `buf`
    auto stage = [&](int buf, int kb) {
#pragma unroll
        for (int it = 0; it < 2; ++it) {           // A: 512 lane-chunks of 16B
            int idx = tid + it * 256;
            int r = idx >> 2, cg = (idx & 3) * 8;
            async_b128(asBase + (unsigned)(buf * ASZ + r * PAD + cg) * 2u,
                       A + (size_t)(bm + r) * K + kb + cg);
        }
#pragma unroll
        for (int it = 0; it < 4; ++it) {           // B: 1024 lane-chunks of 16B
            int idx = tid + it * 256;
            int r = idx >> 2, cg = (idx & 3) * 8;
            async_b128(bsBase + (unsigned)(buf * BSZ + r * PAD + cg) * 2u,
                       Bt + (size_t)(bn + r) * K + kb + cg);
        }
    };

    stage(0, 0);                                   // prologue
    for (int kb = 0; kb < K; kb += BK) {
        const int cur = (kb / BK) & 1;
        wait_async0();                             // my async issues landed in LDS
        __syncthreads();                           // everyone's tile `cur` is ready
        if (kb + BK < K) stage(cur ^ 1, kb + BK);  // overlap next tile with compute

        const h16* Ac = As + cur * ASZ;
        const h16* Bc = Bs + cur * BSZ;
        v16h af[4];
#pragma unroll
        for (int mi = 0; mi < 4; ++mi) {
            FU f;
            const uint4* p = (const uint4*)(Ac + (wm + mi * 16 + lh) * PAD + ka8);
            f.u[0] = p[0];                         // K = ka8 .. ka8+7
            f.u[1] = p[2];                         // K = ka8+16 .. ka8+23
            af[mi] = f.v;
        }
#pragma unroll
        for (int ni = 0; ni < 4; ++ni) {
            FU f;
            const uint4* p = (const uint4*)(Bc + (wn + ni * 16 + lh) * PAD + kb16);
            f.u[0] = p[0];                         // K = kb16 .. kb16+7
            f.u[1] = p[2];                         // K = kb16+16 .. kb16+23
            v16h bf = f.v;
#pragma unroll
            for (int mi = 0; mi < 4; ++mi)
                acc[mi][ni] = __builtin_amdgcn_wmma_f32_16x16x32_f16(
                    false, af[mi], false, bf, (short)0, acc[mi][ni], false, false);
        }
        // no trailing barrier needed: next-iteration barrier (after wait_async0)
        // orders reads of `cur` before anyone re-stages into it
    }

    // ---- epilogue: bias (+resid) + activation, scatter per C/D layout ----
#pragma unroll
    for (int mi = 0; mi < 4; ++mi) {
#pragma unroll
        for (int ni = 0; ni < 4; ++ni) {
            int mbase = bm + wm + mi * 16 + (lane >> 4) * 8;
            int n = bn + wn + ni * 16 + lh;
            float bv = bias[n];
#pragma unroll
            for (int v = 0; v < 8; ++v) {
                int m = mbase + v;
                float c = acc[mi][ni][v] + bv;
                if (resid) c += resid[(size_t)m * N + n];
                if (EPI == 1) c = (c > 0.f) ? (c + 1.f) : (expm1f(c) + 1.f); // elu+1
                else if (EPI == 2) c = 0.5f * c * (1.f + erff(c * 0.70710678118654752f));
                C[(size_t)m * N + n] = (OutT)c;
            }
        }
    }
}

// ---------------------------------------------------------------------------
// Chunked causal linear attention. One block per (batch, head); 128 threads,
// thread i owns row i of each chunk. Running state S[64][64], z[64] in LDS.
// Q/K are already feature-mapped (elu+1) by the GEMM epilogue.
// ---------------------------------------------------------------------------
__global__ __launch_bounds__(128) void k_attn(const float* __restrict__ Q,
                                              const float* __restrict__ Kf,
                                              const float* __restrict__ V,
                                              float* __restrict__ O) {
    __shared__ float ks[CH][DH];
    __shared__ float vs[CH][DH];
    __shared__ float S[DH][DH];
    __shared__ float z[DH];

    const int bh = blockIdx.x;
    const int b = bh >> 4, h = bh & 15;
    const int i = threadIdx.x;

    for (int t = i; t < DH * DH; t += CH) (&S[0][0])[t] = 0.f;
    if (i < DH) z[i] = 0.f;
    __syncthreads();

    float qr[DH], outv[DH];
    for (int nc = 0; nc < NCH; ++nc) {
        const int row = nc * CH + i;
        const size_t base = ((size_t)b * LL + row) * DM + (size_t)h * DH;
        for (int d = 0; d < DH; ++d) {
            qr[d]    = Q[base + d];
            ks[i][d] = Kf[base + d];
            vs[i][d] = V[base + d];
        }
        __syncthreads();

        float den = EPS_ATT;
        for (int m = 0; m < DH; ++m) outv[m] = 0.f;

        // inter-chunk: q_i @ S (exclusive prefix state), den += q_i . z
        for (int d = 0; d < DH; ++d) {
            float qd = qr[d];
            den += qd * z[d];
            const float* Sr = S[d];
            for (int m = 0; m < DH; ++m) outv[m] += qd * Sr[m];
        }
        // intra-chunk causal: sum_{j<=i} (q_i.k_j) * v_j ; den += sum_{j<=i} q_i.k_j
        for (int j = 0; j <= i; ++j) {
            float a = 0.f;
            for (int d = 0; d < DH; ++d) a += qr[d] * ks[j][d];
            den += a;
            for (int m = 0; m < DH; ++m) outv[m] += a * vs[j][m];
        }
        float inv = 1.f / den;
        for (int m = 0; m < DH; ++m) O[base + m] = outv[m] * inv;
        __syncthreads();

        // state update: S += K^T V over this chunk ; z += sum_j k_j
        for (int t = i; t < DH * DH; t += CH) {
            int d = t >> 6, m = t & 63;
            float s = 0.f;
            for (int j = 0; j < CH; ++j) s += ks[j][d] * vs[j][m];
            S[d][m] += s;
        }
        if (i < DH) {
            float s = 0.f;
            for (int j = 0; j < CH; ++j) s += ks[j][i];
            z[i] += s;
        }
        __syncthreads();
    }
}

// ---------------------------------------------------------------------------
// LayerNorm over last dim (1024). One block per row; optional L-flip on write.
// ---------------------------------------------------------------------------
__global__ __launch_bounds__(256) void k_ln(const float* __restrict__ Xin,
                                            const float* __restrict__ g,
                                            const float* __restrict__ bta,
                                            float* __restrict__ Xout, int flip) {
    __shared__ float r1[256], r2[256];
    const int row = blockIdx.x, tid = threadIdx.x;
    const float* xr = Xin + (size_t)row * DM;
    float xv[4];
    float s = 0.f, s2 = 0.f;
#pragma unroll
    for (int k = 0; k < 4; ++k) {
        float v = xr[tid + k * 256];
        xv[k] = v; s += v; s2 += v * v;
    }
    r1[tid] = s; r2[tid] = s2;
    __syncthreads();
    for (int st = 128; st > 0; st >>= 1) {
        if (tid < st) { r1[tid] += r1[tid + st]; r2[tid] += r2[tid + st]; }
        __syncthreads();
    }
    float mu  = r1[0] * (1.f / DM);
    float var = r2[0] * (1.f / DM) - mu * mu;
    float rs  = rsqrtf(var + LN_EPS);
    int orow = row;
    if (flip) { int b = row / LL, l = row % LL; orow = b * LL + (LL - 1 - l); }
    float* yr = Xout + (size_t)orow * DM;
#pragma unroll
    for (int k = 0; k < 4; ++k) {
        int c = tid + k * 256;
        yr[c] = (xv[k] - mu) * rs * g[c] + bta[c];
    }
}

// ---------------------------------------------------------------------------
// Orchestration
// ---------------------------------------------------------------------------
extern "C" void kernel_launch(void* const* d_in, const int* in_sizes, int n_in,
                              void* d_out, int out_size, void* d_ws, size_t ws_size,
                              hipStream_t stream) {
    (void)in_sizes; (void)n_in; (void)out_size; (void)ws_size;
    const float* x   = (const float*)d_in[0];
    const float* Wq  = (const float*)d_in[1];
    const float* bq  = (const float*)d_in[2];
    const float* Wk  = (const float*)d_in[3];
    const float* bk  = (const float*)d_in[4];
    const float* Wv  = (const float*)d_in[5];
    const float* bv  = (const float*)d_in[6];
    const float* Wo  = (const float*)d_in[7];
    const float* bo  = (const float*)d_in[8];
    const float* g1  = (const float*)d_in[9];
    const float* be1 = (const float*)d_in[10];
    const float* W1  = (const float*)d_in[11];
    const float* b1  = (const float*)d_in[12];
    const float* W2  = (const float*)d_in[13];
    const float* b2  = (const float*)d_in[14];
    const float* g2  = (const float*)d_in[15];
    const float* be2 = (const float*)d_in[16];
    const float* gf  = (const float*)d_in[17];
    const float* bf  = (const float*)d_in[18];

    char* ws = (char*)d_ws;
    size_t off = 0;
    auto alloc = [&](size_t bytes) -> void* {
        void* p = ws + off;
        off += (bytes + 255) & ~(size_t)255;
        return p;
    };
    float* X   = (float*)alloc((size_t)MROWS * DM * 4);   // activations (L-flipped)
    h16*   Xh  = (h16*)  alloc((size_t)MROWS * DM * 2);   // f16 staging (reused)
    float* Qb  = (float*)alloc((size_t)MROWS * DM * 4);
    float* Kb  = (float*)alloc((size_t)MROWS * DM * 4);
    float* Vb  = (float*)alloc((size_t)MROWS * DM * 4);
    float* T0  = (float*)alloc((size_t)MROWS * DM * 4);   // attn out / gemm out temp
    h16*   H1h = (h16*)  alloc((size_t)MROWS * DFF * 2);  // gelu(FFN1) in f16
    h16*   Wth = (h16*)  alloc((size_t)DM * DFF * 2);     // transposed f16 weight staging

    const dim3 gWT_sq(DM / 32, DM / 32), bWT(32, 8);
    const dim3 gG_dm(DM / 256, MROWS / 128);
    const dim3 gG_ff(DFF / 256, MROWS / 128);
    const int nElem = MROWS * DM;
    const int cvtBlocks = nElem / (256 * 8);

    k_flip_copy<<<MROWS, 256, 0, stream>>>(x, X);

    for (int l = 0; l < NL; ++l) {
        k_f2h<<<cvtBlocks, 256, 0, stream>>>(X, Xh, nElem);

        // Q = feat(x Wq + bq), K = feat(x Wk + bk), V = x Wv + bv
        k_wt<<<gWT_sq, bWT, 0, stream>>>(Wq + (size_t)l * DM * DM, Wth, DM, DM);
        k_gemm<1, float><<<gG_dm, 256, 0, stream>>>(Xh, Wth, bq + l * DM, nullptr, Qb, MROWS, DM, DM);
        k_wt<<<gWT_sq, bWT, 0, stream>>>(Wk + (size_t)l * DM * DM, Wth, DM, DM);
        k_gemm<1, float><<<gG_dm, 256, 0, stream>>>(Xh, Wth, bk + l * DM, nullptr, Kb, MROWS, DM, DM);
        k_wt<<<gWT_sq, bWT, 0, stream>>>(Wv + (size_t)l * DM * DM, Wth, DM, DM);
        k_gemm<0, float><<<gG_dm, 256, 0, stream>>>(Xh, Wth, bv + l * DM, nullptr, Vb, MROWS, DM, DM);

        // chunked causal linear attention -> T0
        k_attn<<<BB * NH, CH, 0, stream>>>(Qb, Kb, Vb, T0);

        // x = x + attn @ Wo + bo ; then LN1 -> X
        k_f2h<<<cvtBlocks, 256, 0, stream>>>(T0, Xh, nElem);
        k_wt<<<gWT_sq, bWT, 0, stream>>>(Wo + (size_t)l * DM * DM, Wth, DM, DM);
        k_gemm<0, float><<<gG_dm, 256, 0, stream>>>(Xh, Wth, bo + l * DM, X, T0, MROWS, DM, DM);
        k_ln<<<MROWS, 256, 0, stream>>>(T0, g1 + l * DM, be1 + l * DM, X, 0);

        // FFN: h = gelu(X W1 + b1) (f16), y = h W2 + b2 + X ; LN2 -> X
        k_f2h<<<cvtBlocks, 256, 0, stream>>>(X, Xh, nElem);
        k_wt<<<dim3(DFF / 32, DM / 32), bWT, 0, stream>>>(W1 + (size_t)l * DM * DFF, Wth, DM, DFF);
        k_gemm<2, h16><<<gG_ff, 256, 0, stream>>>(Xh, Wth, b1 + l * DFF, nullptr, H1h, MROWS, DFF, DM);
        k_wt<<<dim3(DM / 32, DFF / 32), bWT, 0, stream>>>(W2 + (size_t)l * DFF * DM, Wth, DFF, DM);
        k_gemm<0, float><<<gG_dm, 256, 0, stream>>>(H1h, Wth, b2 + l * DM, X, T0, MROWS, DM, DFF);
        k_ln<<<MROWS, 256, 0, stream>>>(T0, g2 + l * DM, be2 + l * DM, X, 0);
    }

    // final LN with fused un-flip directly into d_out
    k_ln<<<MROWS, 256, 0, stream>>>(X, gf, bf, (float*)d_out, 1);
}
